// WordInQuestionW_36850819399897
// MI455X (gfx1250) — matrix operand order repaired and checked by reading, not compiled, not run on hardware
//
#include <hip/hip_runtime.h>
#include <hip/hip_bf16.h>

// ---------------------------------------------------------------------------
// WordInQuestionW for MI455X (gfx1250, wave32, WMMA)
// sim = (ctx*w) @ q^T  -> masked softmax over T -> sum over j
// bf16 hi/lo split (x = hi+lo), 3x v_wmma_f32_16x16x32_bf16 per K=32 chunk
// gives ~fp32 accuracy at bf16 matrix rates.
// Pass 1: online softmax stats (M_j, S_j) per (b,j) -> d_ws (64 KB).
// Pass 2: recompute sim, emit out[b,t] = sum_j exp(sim-M_j)/S_j.
// This revision: float4/global_load_b128 staging + packed v4bf LDS stores,
// and double-buffered LDS K-chunks (1 barrier per chunk, global loads for
// chunk k+1 overlap WMMA on chunk k).
// ---------------------------------------------------------------------------

#define B_ 64
#define T_ 2048
#define J_ 128
#define D_ 512

constexpr int JT = 32;      // j columns per workgroup (kernel 1)
constexpr int TT = 64;      // t rows per super-tile / per workgroup (kernel 2)
constexpr int DK = 64;      // d (K) chunk staged in LDS
constexpr int NK = D_ / DK; // 8 chunks
constexpr int CP = DK + 8;  // ctx LDS row stride (bf16): 144 B rows, 16B aligned
constexpr int QP = D_ + 8;  // q LDS row stride (kernel 1): 1040 B, 16B aligned
#define NEG_BIG (-3.0e38f)

typedef __attribute__((ext_vector_type(16))) __bf16 v16bf;
typedef __attribute__((ext_vector_type(8)))  __bf16 v8bf;
typedef __attribute__((ext_vector_type(4)))  __bf16 v4bf;
typedef __attribute__((ext_vector_type(8)))  float  v8f;

// Build a v16bf A/B fragment from two 16-byte LDS chunks (ds_load_b128 each).
__device__ __forceinline__ v16bf ld16(const __bf16* p0, const __bf16* p1) {
  v8bf a = *(const v8bf*)p0;
  v8bf b = *(const v8bf*)p1;
  return __builtin_shufflevector(a, b, 0,1,2,3,4,5,6,7,8,9,10,11,12,13,14,15);
}

// hi*hi + hi*lo + lo*hi  (drop lo*lo: ~2^-16 relative error, fp32-class)
__device__ __forceinline__ v8f wmma3(v16bf Ahi, v16bf Alo, v16bf Bhi, v16bf Blo, v8f c) {
  c = __builtin_amdgcn_wmma_f32_16x16x32_bf16(false, Ahi, false, Bhi, (short)0, c, false, false);
  c = __builtin_amdgcn_wmma_f32_16x16x32_bf16(false, Ahi, false, Blo, (short)0, c, false, false);
  c = __builtin_amdgcn_wmma_f32_16x16x32_bf16(false, Alo, false, Bhi, (short)0, c, false, false);
  return c;
}

// Split a float4 into packed bf16 hi/lo quads (8-byte LDS stores).
__device__ __forceinline__ void split4(float4 x, v4bf* hi, v4bf* lo) {
  v4bf h, l;
  h[0] = (__bf16)x.x; l[0] = (__bf16)(x.x - (float)h[0]);
  h[1] = (__bf16)x.y; l[1] = (__bf16)(x.y - (float)h[1]);
  h[2] = (__bf16)x.z; l[2] = (__bf16)(x.z - (float)h[2]);
  h[3] = (__bf16)x.w; l[3] = (__bf16)(x.w - (float)h[3]);
  *hi = h; *lo = l;
}

// ---------------------------------------------------------------------------
// Kernel 1: per (b, j-block of 32) -> column softmax stats over T
// 8 waves = 4 t-subtiles x 2 j-subtiles. Online max/sum across T super-tiles.
// ---------------------------------------------------------------------------
__global__ __launch_bounds__(256)
void wiq_stats_kernel(const float* __restrict__ question,
                      const float* __restrict__ context,
                      const int*   __restrict__ clen,
                      const float* __restrict__ weight,
                      float* __restrict__ Mws, float* __restrict__ Sws) {
  __shared__ __bf16 qhi[JT][QP];
  __shared__ __bf16 qlo[JT][QP];
  __shared__ __bf16 chi[2][TT][CP];   // double-buffered ctx chunk
  __shared__ __bf16 clo[2][TT][CP];
  __shared__ float  redM[8][16];
  __shared__ float  redS[8][16];

  const int b  = blockIdx.y;
  const int j0 = blockIdx.x * JT;
  const int cl = clen[b];

  const int tid  = threadIdx.x;
  const int wave = tid >> 5, lane = tid & 31;
  const int tt   = wave & 3, jj = wave >> 2;      // 4 x 2 wave grid
  const int half = lane >> 4, idx = lane & 15;

  // float4 staging decomposition (fixed per thread)
  const int sr = tid >> 4;            // 0..15 (chunk row group)
  const int sc = (tid & 15) * 4;      // 0..60 (chunk col, float4 granule)

  const float* qb = question + ((size_t)b * J_ + j0) * D_;
  const float* cb = context + (size_t)b * T_ * D_;

  // Stage question block * w (hi/lo bf16), once. float4 loads, v4bf stores.
  {
    const int qd  = (tid & 127) * 4;  // 0..508
    const int qj0 = tid >> 7;         // 0..1
    float4 wv = *(const float4*)(weight + qd);
#pragma unroll
    for (int i = 0; i < 16; ++i) {
      int j = qj0 + i * 2;
      float4 x = *(const float4*)(qb + (size_t)j * D_ + qd);
      x.x *= wv.x; x.y *= wv.y; x.z *= wv.z; x.w *= wv.w;
      split4(x, (v4bf*)&qhi[j][qd], (v4bf*)&qlo[j][qd]);
    }
  }

  float M = NEG_BIG, S = 0.f;
  const int nSup = (cl + TT - 1) / TT;            // skip fully masked tail

  for (int sup = 0; sup < nSup; ++sup) {
    v8f acc = {};
    const int tBase = sup * TT;

    // prologue: stage chunk 0 into buffer 0
#pragma unroll
    for (int i = 0; i < 4; ++i) {
      int r = sr + i * 16;
      float4 x = *(const float4*)(cb + (size_t)(tBase + r) * D_ + sc);
      split4(x, (v4bf*)&chi[0][r][sc], (v4bf*)&clo[0][r][sc]);
    }
    __syncthreads();

    for (int ki = 0; ki < NK; ++ki) {
      const int kc = ki * DK;
      const int cur = ki & 1, nxt = cur ^ 1;
      if (ki + 1 < NK) {      // overlap: stage chunk ki+1 while computing ki
#pragma unroll
        for (int i = 0; i < 4; ++i) {
          int r = sr + i * 16;
          float4 x = *(const float4*)(cb + (size_t)(tBase + r) * D_ + kc + DK + sc);
          split4(x, (v4bf*)&chi[nxt][r][sc], (v4bf*)&clo[nxt][r][sc]);
        }
      }
#pragma unroll
      for (int ks = 0; ks < 2; ++ks) {
        const int k0 = ks * 32;
        const __bf16* ar  = &chi[cur][tt * 16 + idx][0];
        const __bf16* arl = &clo[cur][tt * 16 + idx][0];
        const int a0 = k0 + half * 8, a1 = k0 + 16 + half * 8;
        v16bf Ahi = ld16(ar + a0, ar + a1);
        v16bf Alo = ld16(arl + a0, arl + a1);
        const __bf16* br  = &qhi[jj * 16 + idx][0];
        const __bf16* brl = &qlo[jj * 16 + idx][0];
        const int b0 = kc + k0 + half * 8;
        v16bf Bhi = ld16(br + b0, br + b0 + 16);
        v16bf Blo = ld16(brl + b0, brl + b0 + 16);
        acc = wmma3(Ahi, Alo, Bhi, Blo, acc);
      }
      __syncthreads();
    }

    // Column-wise tile stats (column n lives in lanes n and n+16).
    float vals[8];
    float m = NEG_BIG;
#pragma unroll
    for (int r = 0; r < 8; ++r) {
      int t = tBase + tt * 16 + half * 8 + r;
      float v = (t < cl) ? acc[r] : NEG_BIG;
      vals[r] = v;
      m = fmaxf(m, v);
    }
    m = fmaxf(m, __shfl_xor(m, 16, 32));
    float s = 0.f;
#pragma unroll
    for (int r = 0; r < 8; ++r) {
      int t = tBase + tt * 16 + half * 8 + r;
      if (t < cl) s += __expf(vals[r] - m);
    }
    s += __shfl_xor(s, 16, 32);
    float nM = fmaxf(M, m);
    S = S * __expf(M - nM) + s * __expf(m - nM);
    M = nM;
  }

  // Combine the 4 t-subtile waves per j column.
  if (half == 0) { redM[wave][idx] = M; redS[wave][idx] = S; }
  __syncthreads();
  if (tid < 32) {
    int jjc = tid >> 4, n = tid & 15;
    float Mf = NEG_BIG, Sf = 0.f;
#pragma unroll
    for (int t4 = 0; t4 < 4; ++t4) Mf = fmaxf(Mf, redM[jjc * 4 + t4][n]);
#pragma unroll
    for (int t4 = 0; t4 < 4; ++t4)
      Sf += redS[jjc * 4 + t4][n] * __expf(redM[jjc * 4 + t4][n] - Mf);
    if (!(Sf > 0.f)) { Mf = 0.f; Sf = 1.f; }   // fully masked column
    int j = j0 + jjc * 16 + n;
    Mws[b * J_ + j] = Mf;
    Sws[b * J_ + j] = Sf;
  }
}

// ---------------------------------------------------------------------------
// Kernel 2: per (b, t-block of 64) -> out[b,t] = sum_j exp(sim-M_j)/S_j
// 8 waves = 4 t-subtiles x 2 j-halves; each wave holds 4 accumulator tiles
// (16t x 64j). Double-buffered K chunks for ctx (64xDK) and all 128 q rows.
// ---------------------------------------------------------------------------
__global__ __launch_bounds__(256)
void wiq_out_kernel(const float* __restrict__ question,
                    const float* __restrict__ context,
                    const int*   __restrict__ qlen,
                    const int*   __restrict__ clen,
                    const float* __restrict__ weight,
                    const float* __restrict__ Mws,
                    const float* __restrict__ Sws,
                    float* __restrict__ out) {
  __shared__ __bf16 qhi[2][J_][CP];
  __shared__ __bf16 qlo[2][J_][CP];
  __shared__ __bf16 chi[2][TT][CP];
  __shared__ __bf16 clo[2][TT][CP];
  __shared__ float  Ms[J_], Rs[J_];
  __shared__ float  outRed[2][TT];

  const int b  = blockIdx.y;
  const int t0 = blockIdx.x * TT;
  const int cl = clen[b], ql = qlen[b];
  const int tid = threadIdx.x;

  if (t0 >= cl) {                 // fully masked rows -> zeros
    if (tid < TT) out[(size_t)b * T_ + t0 + tid] = 0.f;
    return;
  }
  if (tid < J_) {
    Ms[tid] = Mws[b * J_ + tid];
    Rs[tid] = 1.f / Sws[b * J_ + tid];
  }

  const int wave = tid >> 5, lane = tid & 31;
  const int tt   = wave & 3, jjH = wave >> 2;   // jjH covers 64 j columns
  const int half = lane >> 4, idx = lane & 15;

  const int sr = tid >> 4;            // 0..15
  const int sc = (tid & 15) * 4;      // 0..60

  v8f acc[4] = {};
  const float* cb = context + ((size_t)b * T_ + t0) * D_;
  const float* qb = question + (size_t)b * J_ * D_;

  // stage chunk kc into buffer buf (ctx 64xDK + q 128xDK, with weight)
  auto stage = [&](int buf, int kc) {
    float4 wv = *(const float4*)(weight + kc + sc);
#pragma unroll
    for (int i = 0; i < 4; ++i) {
      int r = sr + i * 16;
      float4 x = *(const float4*)(cb + (size_t)r * D_ + kc + sc);
      split4(x, (v4bf*)&chi[buf][r][sc], (v4bf*)&clo[buf][r][sc]);
    }
#pragma unroll
    for (int i = 0; i < 8; ++i) {
      int j = sr + i * 16;
      float4 x = *(const float4*)(qb + (size_t)j * D_ + kc + sc);
      x.x *= wv.x; x.y *= wv.y; x.z *= wv.z; x.w *= wv.w;
      split4(x, (v4bf*)&qhi[buf][j][sc], (v4bf*)&qlo[buf][j][sc]);
    }
  };

  stage(0, 0);
  __syncthreads();

  for (int ki = 0; ki < NK; ++ki) {
    const int cur = ki & 1, nxt = cur ^ 1;
    if (ki + 1 < NK) stage(nxt, (ki + 1) * DK);   // overlap with WMMA below
#pragma unroll
    for (int ks = 0; ks < 2; ++ks) {
      const int k0 = ks * 32;
      const __bf16* ar  = &chi[cur][tt * 16 + idx][0];
      const __bf16* arl = &clo[cur][tt * 16 + idx][0];
      const int a0 = k0 + half * 8, a1 = k0 + 16 + half * 8;
      v16bf Ahi = ld16(ar + a0, ar + a1);
      v16bf Alo = ld16(arl + a0, arl + a1);
      const int b0 = k0 + half * 8;
#pragma unroll
      for (int s = 0; s < 4; ++s) {
        const __bf16* br  = &qhi[cur][jjH * 64 + s * 16 + idx][0];
        const __bf16* brl = &qlo[cur][jjH * 64 + s * 16 + idx][0];
        v16bf Bhi = ld16(br + b0, br + b0 + 16);
        v16bf Blo = ld16(brl + b0, brl + b0 + 16);
        acc[s] = wmma3(Ahi, Alo, Bhi, Blo, acc[s]);
      }
    }
    __syncthreads();
  }

  // Epilogue: exp(sim-M)*1/S, masked, then reduce over j.
  float rowAcc[8];
#pragma unroll
  for (int r = 0; r < 8; ++r) rowAcc[r] = 0.f;
#pragma unroll
  for (int s = 0; s < 4; ++s) {
    int j = jjH * 64 + s * 16 + idx;
    bool jok = (j < ql);
    float Mj = Ms[j], Rj = Rs[j];
#pragma unroll
    for (int r = 0; r < 8; ++r) {
      int t = t0 + tt * 16 + half * 8 + r;
      if (jok && t < cl) rowAcc[r] += __expf(acc[s][r] - Mj) * Rj;
    }
  }
#pragma unroll
  for (int off = 1; off < 16; off <<= 1)
#pragma unroll
    for (int r = 0; r < 8; ++r) rowAcc[r] += __shfl_xor(rowAcc[r], off, 32);
  if (idx == 0) {
#pragma unroll
    for (int r = 0; r < 8; ++r) outRed[jjH][tt * 16 + half * 8 + r] = rowAcc[r];
  }
  __syncthreads();
  if (tid < TT) {
    int t = t0 + tid;
    out[(size_t)b * T_ + t] = (t < cl) ? (outRed[0][tid] + outRed[1][tid]) : 0.f;
  }
}

// ---------------------------------------------------------------------------
extern "C" void kernel_launch(void* const* d_in, const int* in_sizes, int n_in,
                              void* d_out, int out_size, void* d_ws, size_t ws_size,
                              hipStream_t stream) {
  const float* question = (const float*)d_in[0];
  const float* context  = (const float*)d_in[1];
  const int*   qlen     = (const int*)d_in[2];
  const int*   clen     = (const int*)d_in[3];
  const float* weight   = (const float*)d_in[4];
  float* out = (float*)d_out;

  float* Mws = (float*)d_ws;            // B*J floats
  float* Sws = Mws + B_ * J_;           // B*J floats (64 KB total)

  dim3 g1(J_ / JT, B_);                 // 4 x 64 = 256 workgroups
  wiq_stats_kernel<<<g1, 256, 0, stream>>>(question, context, clen, weight, Mws, Sws);

  dim3 g2(T_ / TT, B_);                 // 32 x 64 = 2048 workgroups
  wiq_out_kernel<<<g2, 256, 0, stream>>>(question, context, qlen, clen, weight,
                                         Mws, Sws, out);
}
